// JIHLImputer_47004122087476
// MI455X (gfx1250) — matrix operand order repaired and checked by reading, not compiled, not run on hardware
//
#include <hip/hip_runtime.h>
#include <hip/hip_bf16.h>

typedef __attribute__((ext_vector_type(16))) _Float16 v16h;
typedef __attribute__((ext_vector_type(8)))  _Float16 v8h;
typedef __attribute__((ext_vector_type(8)))  float    v8f;

#define EMA_A 0.9f
#define EMA_B 0.1f
#define EPS   1e-8f

// ---- LDS layout (dynamic shared) ----
// sW1: 64x128 f16 swizzled   = 8192 halves  (16KB)
// sW2: 128x128 f16 swizzled  = 16384 halves (32KB)
// sW3: 128x64 f16 swizzled   = 8192 halves  (16KB)
// sB1/sB2/sB3 f32            = 320 floats   (1.25KB)
// sX : 8 waves x 16x64 f32   = 8192 floats  (32KB)
// sH : 8 waves x 16x128 f16  = 16384 halves (32KB)  (reused A->h1->h2)
// sCmp: 8 waves x 4 uint
static constexpr size_t SM_W1   = 0;
static constexpr size_t SM_W2   = SM_W1 + 8192 * 2;
static constexpr size_t SM_W3   = SM_W2 + 16384 * 2;
static constexpr size_t SM_B1   = SM_W3 + 8192 * 2;
static constexpr size_t SM_B2   = SM_B1 + 128 * 4;
static constexpr size_t SM_B3   = SM_B2 + 128 * 4;
static constexpr size_t SM_X    = SM_B3 + 64 * 4;
static constexpr size_t SM_H    = SM_X + 8 * 1024 * 4;
static constexpr size_t SM_CMP  = SM_H + 8 * 2048 * 2;
static constexpr size_t SM_TOTAL = SM_CMP + 32 * 4;

// swizzled B-fragment index: chunk(kt,nt) of 512 halves, lane-major, 16 contiguous halves/lane
__device__ __forceinline__ int bswz(int k, int n, int ntiles) {
    int kt = k >> 5, nt = n >> 4, kin = k & 31, nin = n & 15;
    int chunk = kt * ntiles + nt;
    int L = ((kin >> 4) << 4) | nin;   // lane
    int j = kin & 15;                  // within-lane half
    return chunk * 512 + L * 16 + j;
}

// A fragment 16x32 f16 from row-major LDS buffer (two ds_load_b128 per lane)
__device__ __forceinline__ v16h load_a_frag(const _Float16* buf, int ld, int kbase, int lane) {
    int m = lane & 15, hi = (lane >> 4) & 1;
    const v8h* p0 = (const v8h*)(buf + m * ld + kbase + hi * 16);
    const v8h* p1 = (const v8h*)(buf + m * ld + kbase + hi * 16 + 8);
    v8h lo = *p0, hv = *p1;
    v16h a;
#pragma unroll
    for (int i = 0; i < 8; ++i) { a[i] = lo[i]; a[i + 8] = hv[i]; }
    return a;
}

// B fragment 32x16 f16 from swizzled LDS weights (contiguous 32B per lane)
__device__ __forceinline__ v16h load_b_frag(const _Float16* base, int chunk, int lane) {
    return *(const v16h*)(base + chunk * 512 + lane * 16);
}

template <int KT, int NT>
__device__ __forceinline__ void gemm_tiles(const _Float16* Abuf, int lda,
                                           const _Float16* Bbase, int lane, v8f* acc) {
#pragma unroll
    for (int nt = 0; nt < NT; ++nt) { v8f z = {}; acc[nt] = z; }
#pragma unroll
    for (int kt = 0; kt < KT; ++kt) {
        v16h a = load_a_frag(Abuf, lda, kt * 32, lane);
#pragma unroll
        for (int nt = 0; nt < NT; ++nt) {
            v16h b = load_b_frag(Bbase, kt * NT + nt, lane);
            acc[nt] = __builtin_amdgcn_wmma_f32_16x16x32_f16(
                false, a, false, b, (short)0, acc[nt], false, false);
        }
    }
}

// C/D f32 16x16 tile -> bias + relu -> f16 row-major LDS
template <int NT>
__device__ __forceinline__ void store_relu(const v8f* acc, const float* bias,
                                           _Float16* dst, int ld, int lane) {
    int n = lane & 15, rb = ((lane >> 4) & 1) * 8;
#pragma unroll
    for (int nt = 0; nt < NT; ++nt) {
#pragma unroll
        for (int p = 0; p < 8; ++p) {
            int row = p + rb, col = nt * 16 + n;
            float v = acc[nt][p] + bias[col];
            v = fmaxf(v, 0.f);
            dst[row * ld + col] = (_Float16)v;
        }
    }
}

__device__ __forceinline__ float dot4(const float4& a, const float4& b) {
    return a.x * b.x + a.y * b.y + a.z * b.z + a.w * b.w;
}
__device__ __forceinline__ float dget(const float* d, int k, int l) {
    int a = k < l ? k : l, b = k < l ? l : k;
    const int base[4] = {0, 4, 7, 9};
    return d[base[a] + (b - a)];
}

extern "C" __global__ __launch_bounds__(256, 1)
void jihl_imputer_kernel(const float* __restrict__ X, const int* __restrict__ Mk,
                         const float* __restrict__ Hv,
                         const float* __restrict__ W1, const float* __restrict__ B1,
                         const float* __restrict__ W2, const float* __restrict__ B2,
                         const float* __restrict__ W3, const float* __restrict__ B3,
                         float* __restrict__ Out, int N, int G) {
    extern __shared__ __attribute__((aligned(128))) char smem[];
    _Float16* sW1 = (_Float16*)(smem + SM_W1);
    _Float16* sW2 = (_Float16*)(smem + SM_W2);
    _Float16* sW3 = (_Float16*)(smem + SM_W3);
    float* sB1 = (float*)(smem + SM_B1);
    float* sB2 = (float*)(smem + SM_B2);
    float* sB3 = (float*)(smem + SM_B3);
    float* sX  = (float*)(smem + SM_X);
    _Float16* sH = (_Float16*)(smem + SM_H);
    unsigned* sCmp = (unsigned*)(smem + SM_CMP);

    const int tid = threadIdx.x;
    // ---- stage weights once per block, pre-swizzled to B-fragment order ----
    for (int e = tid; e < 8192; e += 256) {            // W1 effective: rows 0..63 of [192,128]
        int k = e >> 7, n = e & 127;
        sW1[bswz(k, n, 8)] = (_Float16)W1[e];
    }
    for (int e = tid; e < 16384; e += 256) {           // W2 [128,128]
        int k = e >> 7, n = e & 127;
        sW2[bswz(k, n, 8)] = (_Float16)W2[e];
    }
    for (int e = tid; e < 8192; e += 256) {            // W3 [128,64]
        int k = e >> 6, n = e & 63;
        sW3[bswz(k, n, 4)] = (_Float16)W3[e];
    }
    if (tid < 128) { sB1[tid] = B1[tid]; sB2[tid] = B2[tid]; }
    if (tid < 64) sB3[tid] = B3[tid];
    __syncthreads();

    const int lane = tid & 31;
    const int wv = tid >> 5;
    const int g = lane >> 3;    // sample-in-group 0..3
    const int j = lane & 7;     // lane-in-sample-group
    _Float16* sHw = sH + wv * 2048;
    float* sXw = sX + wv * 1024;

    const int waveGlobal = blockIdx.x * 8 + wv;
    const int stride = gridDim.x * 8;

    for (int gp = waveGlobal; gp < G; gp += stride) {
        const size_t samp = (size_t)gp * 4 + g;

        // prefetch next group's h_views/X
        if (gp + stride < G) {
            size_t ns = (size_t)(gp + stride) * 4;
            __builtin_prefetch((const char*)(Hv + ns * 512) + lane * 256, 0, 1);
            __builtin_prefetch((const char*)(X + ns * 256) + lane * 128, 0, 1);
        }

        // ---- 10 unique Gram dots of h_views[samp] (8-lane cooperative) ----
        float d10[10];
#pragma unroll
        for (int i = 0; i < 10; ++i) d10[i] = 0.f;
#pragma unroll
        for (int t = 0; t < 4; ++t) {
            float4 hv4[4];
#pragma unroll
            for (int k = 0; k < 4; ++k)
                hv4[k] = *(const float4*)(Hv + samp * 512 + k * 128 + t * 32 + j * 4);
            int idx = 0;
#pragma unroll
            for (int k = 0; k < 4; ++k)
#pragma unroll
                for (int l = k; l < 4; ++l) d10[idx++] += dot4(hv4[k], hv4[l]);
        }
#pragma unroll
        for (int m = 1; m < 8; m <<= 1)
#pragma unroll
            for (int i = 0; i < 10; ++i) d10[i] += __shfl_xor(d10[i], m, 32);

        // ---- completeness (mask is a VW-repeat: one int per view suffices) ----
        bool comp[4];
#pragma unroll
        for (int l = 0; l < 4; ++l) comp[l] = Mk[samp * 256 + l * 64] != 0;
        if (j == 0)
            sCmp[wv * 4 + g] = (comp[0] ? 1u : 0u) | (comp[1] ? 2u : 0u) |
                               (comp[2] ? 4u : 0u) | (comp[3] ? 8u : 0u);

        // ---- confidence / conflict / softmax alpha (redundant per lane) ----
        float nrm[4];
#pragma unroll
        for (int k = 0; k < 4; ++k) nrm[k] = fmaxf(sqrtf(fmaxf(dget(d10, k, k), 0.f)), EPS);
        float conf[4], confl[4];
        bool hasv[4];
#pragma unroll
        for (int k = 0; k < 4; ++k) {
            int na = 0;
            float sc = 0.f, sdk = 0.f;
#pragma unroll
            for (int l = 0; l < 4; ++l)
                if (comp[l] && l != k) {
                    ++na;
                    sc += dget(d10, k, l) / (nrm[k] * nrm[l]);
                    sdk += dget(d10, k, l);
                }
            hasv[k] = na > 0;
            float denom = (float)(na > 1 ? na : 1);
            conf[k] = (comp[k] && hasv[k]) ? 1.f / (1.f + expf(-sc / denom)) : 0.5f;
            float hm2 = 0.f;
#pragma unroll
            for (int l = 0; l < 4; ++l)
                if (comp[l] && l != k)
#pragma unroll
                    for (int lp = 0; lp < 4; ++lp)
                        if (comp[lp] && lp != k) hm2 += dget(d10, l, lp);
            hm2 /= denom * denom;
            float hm = fmaxf(sqrtf(fmaxf(hm2, 0.f)), EPS);
            float cos_km = (sdk / denom) / (nrm[k] * hm);
            confl[k] = (comp[k] && hasv[k]) ? (1.f - cos_km) * 0.5f : 0.f;
        }
        float logit[4];
#pragma unroll
        for (int l = 0; l < 4; ++l) logit[l] = conf[l] * (1.f - confl[l]);
        float alpha[4][4];
#pragma unroll
        for (int k = 0; k < 4; ++k) {
            float z[4], zm = -3e9f;
#pragma unroll
            for (int l = 0; l < 4; ++l) {
                z[l] = (comp[l] && l != k) ? logit[l] : -1e9f;
                zm = fmaxf(zm, z[l]);
            }
            float s = 0.f;
#pragma unroll
            for (int l = 0; l < 4; ++l) { alpha[k][l] = expf(z[l] - zm); s += alpha[k][l]; }
            float inv = hasv[k] ? 1.f / s : 0.f;   // no available view -> info_view = 0
#pragma unroll
            for (int l = 0; l < 4; ++l) alpha[k][l] *= inv;
        }

        // ---- info_view -> A-tile (f16), stage X (f32) for EMA ----
#pragma unroll
        for (int t = 0; t < 2; ++t) {
            int col4 = j * 4 + t * 32;
            float4 xl[4];
#pragma unroll
            for (int l = 0; l < 4; ++l) {
                xl[l] = *(const float4*)(X + samp * 256 + l * 64 + col4);
                *(float4*)(sXw + (g * 4 + l) * 64 + col4) = xl[l];
            }
#pragma unroll
            for (int k = 0; k < 4; ++k) {
                float4 iv = {0.f, 0.f, 0.f, 0.f};
#pragma unroll
                for (int l = 0; l < 4; ++l) {
                    iv.x += alpha[k][l] * xl[l].x;
                    iv.y += alpha[k][l] * xl[l].y;
                    iv.z += alpha[k][l] * xl[l].z;
                    iv.w += alpha[k][l] * xl[l].w;
                }
                _Float16* dst = sHw + (g * 4 + k) * 64 + col4;
                dst[0] = (_Float16)iv.x; dst[1] = (_Float16)iv.y;
                dst[2] = (_Float16)iv.z; dst[3] = (_Float16)iv.w;
            }
        }

        // ---- MLP: 16 rows through 3 WMMA GEMMs (buffer recycled A->h1->h2) ----
        v8f acc1[8];
        gemm_tiles<2, 8>(sHw, 64, sW1, lane, acc1);    // [16,64]@[64,128]
        store_relu<8>(acc1, sB1, sHw, 128, lane);
        v8f acc2[8];
        gemm_tiles<4, 8>(sHw, 128, sW2, lane, acc2);   // [16,128]@[128,128]
        store_relu<8>(acc2, sB2, sHw, 128, lane);
        v8f acc3[4];
        gemm_tiles<4, 4>(sHw, 128, sW3, lane, acc3);   // [16,128]@[128,64]

        // ---- bias + EMA blend + store ----
        int n = lane & 15, rb = ((lane >> 4) & 1) * 8;
#pragma unroll
        for (int nt = 0; nt < 4; ++nt) {
#pragma unroll
            for (int p = 0; p < 8; ++p) {
                int row = p + rb, col = nt * 16 + n;
                float pred = acc3[nt][p] + sB3[col];
                float xv = sXw[row * 64 + col];
                bool c = (sCmp[wv * 4 + (row >> 2)] >> (row & 3)) & 1u;
                float o = c ? xv : (EMA_A * xv + EMA_B * pred);
                Out[(size_t)gp * 1024 + row * 64 + col] = o;
            }
        }
    }
}

extern "C" void kernel_launch(void* const* d_in, const int* in_sizes, int n_in,
                              void* d_out, int out_size, void* d_ws, size_t ws_size,
                              hipStream_t stream) {
    (void)n_in; (void)d_ws; (void)ws_size; (void)out_size;
    const float* X  = (const float*)d_in[0];
    const int*   Mk = (const int*)d_in[1];
    const float* Hv = (const float*)d_in[2];
    // d_in[3], d_in[4] (edges) unused, as in the reference
    const float* W1 = (const float*)d_in[5];
    const float* B1 = (const float*)d_in[6];
    const float* W2 = (const float*)d_in[7];
    const float* B2 = (const float*)d_in[8];
    const float* W3 = (const float*)d_in[9];
    const float* B3 = (const float*)d_in[10];
    float* Out = (float*)d_out;

    int N = in_sizes[0] / 256;
    int G = N / 4;
    if (G <= 0) return;
    int blocks = (G + 7) / 8;
    if (blocks > 2048) blocks = 2048;

    static int smem_attr_set = 0;  // idempotent attribute set (not a stream op)
    if (!smem_attr_set) {
        hipFuncSetAttribute((const void*)jihl_imputer_kernel,
                            hipFuncAttributeMaxDynamicSharedMemorySize,
                            (int)SM_TOTAL);
        smem_attr_set = 1;
    }

    hipLaunchKernelGGL(jihl_imputer_kernel, dim3(blocks), dim3(256), SM_TOTAL, stream,
                       X, Mk, Hv, W1, B1, W2, B2, W3, B3, Out, N, G);
}
